// GNN_X_11751030522764
// MI455X (gfx1250) — matrix operand order repaired and checked by reading, not compiled, not run on hardware
//
#include <hip/hip_runtime.h>

typedef __attribute__((ext_vector_type(16))) __bf16 v16bf;
typedef __attribute__((ext_vector_type(8)))  float  v8f;

#define TT 8
#define NN 5000
#define EE 50000
#define HD 64

#define LDS_FENCE() asm volatile("s_wait_dscnt 0x0" ::: "memory")

union Frag {
  unsigned u[8];
  uint4    q[2];
  v16bf    v;
};

__device__ __forceinline__ unsigned short f2bf(float f) {
  union { float f; unsigned u; } c; c.f = f;
  unsigned u = c.u;
  u += 0x7FFFu + ((u >> 16) & 1u);   // round-to-nearest-even
  return (unsigned short)(u >> 16);
}
__device__ __forceinline__ unsigned pack2(float a, float b) {
  return (unsigned)f2bf(a) | ((unsigned)f2bf(b) << 16);
}
__device__ __forceinline__ float siluf(float x) {
  return x / (1.f + __expf(-x));
}
__device__ __forceinline__ v8f zero8() {
  v8f z;
  #pragma unroll
  for (int i = 0; i < 8; ++i) z[i] = 0.f;
  return z;
}

// ---------------------------------------------------------------------------
// Convert fp32 weights (rIn x cIn, row-major, batched) into bf16 WMMA
// B-fragments: dword index = (((b*KC+kc)*NT+nt)*32 + lane)*8 + v
// holding pack(W[k][n], W[k+1][n]) with k = kc*32 + 16*(v>>2) + 8*(lane>>4) + 2*(v&3),
// n = nt*16 + (lane&15).  Out-of-range rows/cols are zero-padded.
// ---------------------------------------------------------------------------
__global__ void conv_wfrag(const float* __restrict__ src, unsigned* __restrict__ dst,
                           int batch, int rIn, int cIn, int KC, int NT) {
  int idx = blockIdx.x * blockDim.x + threadIdx.x;
  int total = batch * KC * NT * 32 * 8;
  if (idx >= total) return;
  int v = idx & 7;
  int lane = (idx >> 3) & 31;
  int rest = idx >> 8;
  int nt = rest % NT; rest /= NT;
  int kc = rest % KC;
  int b  = rest / KC;
  int k = kc * 32 + ((v >> 2) << 4) + ((lane >> 4) << 3) + 2 * (v & 3);
  int n = nt * 16 + (lane & 15);
  const float* S = src + (size_t)b * rIn * cIn;
  float f0 = (k     < rIn && n < cIn) ? S[(size_t)k * cIn + n]       : 0.f;
  float f1 = (k + 1 < rIn && n < cIn) ? S[(size_t)(k + 1) * cIn + n] : 0.f;
  dst[idx] = pack2(f0, f1);
}

// ---------------------------------------------------------------------------
// h = nodes @ emb_w + emb_b + time_emb[t]
// ---------------------------------------------------------------------------
__global__ void embed_kernel(const float* __restrict__ nodes, const float* __restrict__ emb_w,
                             const float* __restrict__ emb_b, const float* __restrict__ te,
                             float* __restrict__ h) {
  int idx = blockIdx.x * blockDim.x + threadIdx.x;
  if (idx >= TT * NN * HD) return;
  int j = idx & 63;
  int tn = idx >> 6;
  int t = tn / NN;
  const float* nr = nodes + (size_t)tn * 6;
  float s = emb_b[j] + te[t * HD + j];
  #pragma unroll
  for (int i = 0; i < 6; ++i) s += nr[i] * emb_w[i * HD + j];
  h[idx] = s;
}

__global__ void zero_kernel(float* __restrict__ p, int n) {
  int i = blockIdx.x * blockDim.x + threadIdx.x;
  if (i < n) p[i] = 0.f;
}

// ---------------------------------------------------------------------------
// Edge kernel: per 16-edge tile gather src/dst/ea -> bf16 A-tile (K padded to
// 160), WMMA chain (silu(mW1+b1), silu(mW2+b2)), scatter-add into agg[row].
// 128 threads = 4 waves, each wave grid-strides over tiles.
// ---------------------------------------------------------------------------
__global__ void __launch_bounds__(128)
edge_kernel(const float* __restrict__ h, const int* __restrict__ edges,
            const float* __restrict__ eattr,
            const unsigned* __restrict__ w1f, const unsigned* __restrict__ w2f,
            const float* __restrict__ b1g, const float* __restrict__ b2g,
            float* __restrict__ agg, int l) {
  __shared__ __align__(16) unsigned sW1[5120];        // 5 kc * 4 nt * 256 dw
  __shared__ __align__(16) unsigned sW2[2048];        // 2 kc * 4 nt * 256 dw
  __shared__ __align__(16) unsigned sA[4 * 1344];     // 4 waves * 16 rows * 84 dw
  __shared__ int sIdx[4 * 32];

  const int tid = threadIdx.x;
  const unsigned* w1l = w1f + (size_t)l * 5120;
  const unsigned* w2l = w2f + (size_t)l * 2048;
  for (int i = tid; i < 1280; i += 128) ((uint4*)sW1)[i] = ((const uint4*)w1l)[i];
  for (int i = tid; i < 512;  i += 128) ((uint4*)sW2)[i] = ((const uint4*)w2l)[i];
  __syncthreads();

  const int wave = tid >> 5, lane = tid & 31;
  const int half = lane >> 4, nlo = lane & 15;
  unsigned* A = sA + wave * 1344;                     // row stride 84 dwords (168 bf16)
  unsigned short* A16 = (unsigned short*)A;
  int* eidx = sIdx + wave * 32;
  const float* b1 = b1g + l * HD;
  const float* b2 = b2g + l * HD;

  const int tilesPerT = EE / 16;                      // 3125
  const int numTiles = TT * tilesPerT;
  const int step = gridDim.x * 4;
  for (int tile = blockIdx.x * 4 + wave; tile < numTiles; tile += step) {
    const int t = tile / tilesPerT;
    const int e0 = (tile - t * tilesPerT) * 16;

    // edge indices: lanes 0-15 -> row[], lanes 16-31 -> col[]
    eidx[lane] = edges[(half ? EE : 0) + e0 + nlo];
    LDS_FENCE();

    // gather: lanes<16 load src row chunk, lanes>=16 dst row chunk
    #pragma unroll 4
    for (int m = 0; m < 16; ++m) {
      const int node = eidx[half * 16 + m];
      float4 x = ((const float4*)(h + ((size_t)t * NN + node) * HD))[nlo];
      unsigned* dst = A + m * 84 + half * 32 + nlo * 2;
      dst[0] = pack2(x.x, x.y);
      dst[1] = pack2(x.z, x.w);
    }
    // ea -> cols 128,129 ; zero-pad cols 130..159
    if (!half) {
      float2 e2 = ((const float2*)eattr)[e0 + nlo];
      A[nlo * 84 + 64] = pack2(e2.x, e2.y);
    } else {
      unsigned* z = A + nlo * 84 + 65;
      #pragma unroll
      for (int i = 0; i < 15; ++i) z[i] = 0u;
    }
    LDS_FENCE();

    // GEMM1: (16 x 160) x (160 x 64)
    v8f acc[4];
    #pragma unroll
    for (int nt = 0; nt < 4; ++nt) acc[nt] = zero8();
    #pragma unroll
    for (int kc = 0; kc < 5; ++kc) {
      Frag a;
      const uint4* ap = (const uint4*)(A + nlo * 84 + kc * 16 + half * 4);
      a.q[0] = ap[0]; a.q[1] = ap[2];
      #pragma unroll
      for (int nt = 0; nt < 4; ++nt) {
        Frag b;
        const uint4* bp = (const uint4*)(sW1 + ((kc * 4 + nt) * 32 + lane) * 8);
        b.q[0] = bp[0]; b.q[1] = bp[1];
        acc[nt] = __builtin_amdgcn_wmma_f32_16x16x32_bf16(false, a.v, false, b.v,
                                                          (short)0, acc[nt], false, false);
      }
    }
    // bias + silu -> A-tile (cols 0..63) as bf16 for GEMM2
    #pragma unroll
    for (int nt = 0; nt < 4; ++nt) {
      const int col = nt * 16 + nlo;
      const float bb = b1[col];
      #pragma unroll
      for (int v = 0; v < 8; ++v)
        A16[(v + 8 * half) * 168 + col] = f2bf(siluf(acc[nt][v] + bb));
    }
    LDS_FENCE();

    // GEMM2: (16 x 64) x (64 x 64)
    #pragma unroll
    for (int nt = 0; nt < 4; ++nt) acc[nt] = zero8();
    #pragma unroll
    for (int kc = 0; kc < 2; ++kc) {
      Frag a;
      const uint4* ap = (const uint4*)(A + nlo * 84 + kc * 16 + half * 4);
      a.q[0] = ap[0]; a.q[1] = ap[2];
      #pragma unroll
      for (int nt = 0; nt < 4; ++nt) {
        Frag b;
        const uint4* bp = (const uint4*)(sW2 + ((kc * 4 + nt) * 32 + lane) * 8);
        b.q[0] = bp[0]; b.q[1] = bp[1];
        acc[nt] = __builtin_amdgcn_wmma_f32_16x16x32_bf16(false, a.v, false, b.v,
                                                          (short)0, acc[nt], false, false);
      }
    }
    // bias + silu + scatter-add into agg[t, row[e]]
    #pragma unroll
    for (int nt = 0; nt < 4; ++nt) {
      const int col = nt * 16 + nlo;
      const float bb = b2[col];
      #pragma unroll
      for (int v = 0; v < 8; ++v) {
        const float val = siluf(acc[nt][v] + bb);
        const int node = eidx[v + 8 * half];
        atomicAdd(agg + ((size_t)t * NN + node) * HD + col, val);
      }
    }
  }
}

// ---------------------------------------------------------------------------
// Node kernel: hn = [h, agg] (K=128) -> silu(hn W1 + b1) W2 + b2 -> h (in place)
// 256 threads = 8 waves.
// ---------------------------------------------------------------------------
__global__ void __launch_bounds__(256)
node_kernel(float* __restrict__ h, const float* __restrict__ agg,
            const unsigned* __restrict__ w1f, const unsigned* __restrict__ w2f,
            const float* __restrict__ b1g, const float* __restrict__ b2g, int l) {
  __shared__ __align__(16) unsigned sW1[4096];        // 4 kc * 4 nt * 256 dw
  __shared__ __align__(16) unsigned sW2[2048];
  __shared__ __align__(16) unsigned sA[8 * 1088];     // 16 rows * 68 dw per wave

  const int tid = threadIdx.x;
  const unsigned* w1l = w1f + (size_t)l * 4096;
  const unsigned* w2l = w2f + (size_t)l * 2048;
  for (int i = tid; i < 1024; i += 256) ((uint4*)sW1)[i] = ((const uint4*)w1l)[i];
  for (int i = tid; i < 512;  i += 256) ((uint4*)sW2)[i] = ((const uint4*)w2l)[i];
  __syncthreads();

  const int wave = tid >> 5, lane = tid & 31;
  const int half = lane >> 4, nlo = lane & 15;
  unsigned* A = sA + wave * 1088;                     // row stride 68 dwords (136 bf16)
  unsigned short* A16 = (unsigned short*)A;
  const float* b1 = b1g + l * HD;
  const float* b2 = b2g + l * HD;

  const int nTilesN = (NN + 15) / 16;                 // 313
  const int numTiles = TT * nTilesN;
  const int step = gridDim.x * 8;
  for (int tile = blockIdx.x * 8 + wave; tile < numTiles; tile += step) {
    const int t = tile / nTilesN;
    const int n0 = (tile - t * nTilesN) * 16;
    LDS_FENCE();
    // lanes<16 load h row chunk (cols 0..63), lanes>=16 load agg chunk (64..127)
    #pragma unroll 4
    for (int m = 0; m < 16; ++m) {
      int n = n0 + m; if (n >= NN) n = NN - 1;
      const float* src = (half ? agg : h) + ((size_t)t * NN + n) * HD;
      float4 x = ((const float4*)src)[nlo];
      unsigned* dst = A + m * 68 + half * 32 + nlo * 2;
      dst[0] = pack2(x.x, x.y);
      dst[1] = pack2(x.z, x.w);
    }
    LDS_FENCE();

    v8f acc[4];
    #pragma unroll
    for (int nt = 0; nt < 4; ++nt) acc[nt] = zero8();
    #pragma unroll
    for (int kc = 0; kc < 4; ++kc) {
      Frag a;
      const uint4* ap = (const uint4*)(A + nlo * 68 + kc * 16 + half * 4);
      a.q[0] = ap[0]; a.q[1] = ap[2];
      #pragma unroll
      for (int nt = 0; nt < 4; ++nt) {
        Frag b;
        const uint4* bp = (const uint4*)(sW1 + ((kc * 4 + nt) * 32 + lane) * 8);
        b.q[0] = bp[0]; b.q[1] = bp[1];
        acc[nt] = __builtin_amdgcn_wmma_f32_16x16x32_bf16(false, a.v, false, b.v,
                                                          (short)0, acc[nt], false, false);
      }
    }
    #pragma unroll
    for (int nt = 0; nt < 4; ++nt) {
      const int col = nt * 16 + nlo;
      const float bb = b1[col];
      #pragma unroll
      for (int v = 0; v < 8; ++v)
        A16[(v + 8 * half) * 136 + col] = f2bf(siluf(acc[nt][v] + bb));
    }
    LDS_FENCE();

    #pragma unroll
    for (int nt = 0; nt < 4; ++nt) acc[nt] = zero8();
    #pragma unroll
    for (int kc = 0; kc < 2; ++kc) {
      Frag a;
      const uint4* ap = (const uint4*)(A + nlo * 68 + kc * 16 + half * 4);
      a.q[0] = ap[0]; a.q[1] = ap[2];
      #pragma unroll
      for (int nt = 0; nt < 4; ++nt) {
        Frag b;
        const uint4* bp = (const uint4*)(sW2 + ((kc * 4 + nt) * 32 + lane) * 8);
        b.q[0] = bp[0]; b.q[1] = bp[1];
        acc[nt] = __builtin_amdgcn_wmma_f32_16x16x32_bf16(false, a.v, false, b.v,
                                                          (short)0, acc[nt], false, false);
      }
    }
    // h = (...) @ W2 + b2   (no activation), write in place
    #pragma unroll
    for (int nt = 0; nt < 4; ++nt) {
      const int col = nt * 16 + nlo;
      const float bb = b2[col];
      #pragma unroll
      for (int v = 0; v < 8; ++v) {
        const int n = n0 + v + 8 * half;
        if (n < NN) h[((size_t)t * NN + n) * HD + col] = acc[nt][v] + bb;
      }
    }
  }
}

// ---------------------------------------------------------------------------
// Decoder: x = silu(h W1 + b1) W2 + b2 (N=12 padded to 16), then
// out[n] = sum_t softmax(theta)[t] * x[t,n]
// ---------------------------------------------------------------------------
__global__ void __launch_bounds__(256)
dec_kernel(const float* __restrict__ h, const unsigned* __restrict__ w1f,
           const unsigned* __restrict__ w2f, const float* __restrict__ b1,
           const float* __restrict__ b2, const float* __restrict__ theta,
           float* __restrict__ out) {
  __shared__ __align__(16) unsigned sW1[2048];        // 2 kc * 4 nt * 256 dw
  __shared__ __align__(16) unsigned sW2[512];         // 2 kc * 1 nt * 256 dw
  __shared__ __align__(16) unsigned sA[8 * 576];      // 16 rows * 36 dw per wave

  const int tid = threadIdx.x;
  for (int i = tid; i < 512; i += 256) ((uint4*)sW1)[i] = ((const uint4*)w1f)[i];
  for (int i = tid; i < 128; i += 256) ((uint4*)sW2)[i] = ((const uint4*)w2f)[i];
  __syncthreads();

  float wgt[TT];
  {
    float mx = -1e30f;
    #pragma unroll
    for (int t = 0; t < TT; ++t) { wgt[t] = theta[t]; mx = fmaxf(mx, wgt[t]); }
    float s = 0.f;
    #pragma unroll
    for (int t = 0; t < TT; ++t) { wgt[t] = __expf(wgt[t] - mx); s += wgt[t]; }
    const float inv = 1.f / s;
    #pragma unroll
    for (int t = 0; t < TT; ++t) wgt[t] *= inv;
  }

  const int wave = tid >> 5, lane = tid & 31;
  const int half = lane >> 4, nlo = lane & 15;
  unsigned* A = sA + wave * 576;                      // row stride 36 dwords (72 bf16)
  unsigned short* A16 = (unsigned short*)A;

  const int nTilesN = (NN + 15) / 16;
  const int step = gridDim.x * 8;
  for (int tile = blockIdx.x * 8 + wave; tile < nTilesN; tile += step) {
    const int n0 = tile * 16;
    v8f accO = zero8();
    for (int t = 0; t < TT; ++t) {
      LDS_FENCE();
      #pragma unroll
      for (int it = 0; it < 8; ++it) {
        const int m = it * 2 + half;
        int n = n0 + m; if (n >= NN) n = NN - 1;
        float4 x = ((const float4*)(h + ((size_t)t * NN + n) * HD))[nlo];
        unsigned* dst = A + m * 36 + nlo * 2;
        dst[0] = pack2(x.x, x.y);
        dst[1] = pack2(x.z, x.w);
      }
      LDS_FENCE();
      v8f acc[4];
      #pragma unroll
      for (int nt = 0; nt < 4; ++nt) acc[nt] = zero8();
      #pragma unroll
      for (int kc = 0; kc < 2; ++kc) {
        Frag a;
        const uint4* ap = (const uint4*)(A + nlo * 36 + kc * 16 + half * 4);
        a.q[0] = ap[0]; a.q[1] = ap[2];
        #pragma unroll
        for (int nt = 0; nt < 4; ++nt) {
          Frag b;
          const uint4* bp = (const uint4*)(sW1 + ((kc * 4 + nt) * 32 + lane) * 8);
          b.q[0] = bp[0]; b.q[1] = bp[1];
          acc[nt] = __builtin_amdgcn_wmma_f32_16x16x32_bf16(false, a.v, false, b.v,
                                                            (short)0, acc[nt], false, false);
        }
      }
      #pragma unroll
      for (int nt = 0; nt < 4; ++nt) {
        const int col = nt * 16 + nlo;
        const float bb = b1[col];
        #pragma unroll
        for (int v = 0; v < 8; ++v)
          A16[(v + 8 * half) * 72 + col] = f2bf(siluf(acc[nt][v] + bb));
      }
      LDS_FENCE();
      v8f y = zero8();
      #pragma unroll
      for (int kc = 0; kc < 2; ++kc) {
        Frag a;
        const uint4* ap = (const uint4*)(A + nlo * 36 + kc * 16 + half * 4);
        a.q[0] = ap[0]; a.q[1] = ap[2];
        Frag b;
        const uint4* bp = (const uint4*)(sW2 + (kc * 32 + lane) * 8);
        b.q[0] = bp[0]; b.q[1] = bp[1];
        y = __builtin_amdgcn_wmma_f32_16x16x32_bf16(false, a.v, false, b.v,
                                                    (short)0, y, false, false);
      }
      const float bb2 = (nlo < 12) ? b2[nlo] : 0.f;
      #pragma unroll
      for (int v = 0; v < 8; ++v) accO[v] += wgt[t] * (y[v] + bb2);
    }
    if (nlo < 12) {
      #pragma unroll
      for (int v = 0; v < 8; ++v) {
        const int n = n0 + v + 8 * half;
        if (n < NN) out[n * 12 + nlo] = accO[v];
      }
    }
  }
}

// ---------------------------------------------------------------------------
extern "C" void kernel_launch(void* const* d_in, const int* in_sizes, int n_in,
                              void* d_out, int out_size, void* d_ws, size_t ws_size,
                              hipStream_t stream) {
  (void)in_sizes; (void)n_in; (void)out_size; (void)ws_size;
  const float* nodes    = (const float*)d_in[0];
  const int*   edges    = (const int*)d_in[1];
  const float* edge_attr= (const float*)d_in[2];
  const float* emb_w    = (const float*)d_in[3];
  const float* emb_b    = (const float*)d_in[4];
  const float* time_emb = (const float*)d_in[5];
  const float* theta    = (const float*)d_in[6];
  const float* edge_w1  = (const float*)d_in[7];
  const float* edge_b1  = (const float*)d_in[8];
  const float* edge_w2  = (const float*)d_in[9];
  const float* edge_b2  = (const float*)d_in[10];
  const float* node_w1  = (const float*)d_in[11];
  const float* node_b1  = (const float*)d_in[12];
  const float* node_w2  = (const float*)d_in[13];
  const float* node_b2  = (const float*)d_in[14];
  const float* dec_w1   = (const float*)d_in[15];
  const float* dec_b1   = (const float*)d_in[16];
  const float* dec_w2   = (const float*)d_in[17];
  const float* dec_b2   = (const float*)d_in[18];

  char* ws = (char*)d_ws;
  size_t off = 0;
  float* h   = (float*)(ws + off); off += (size_t)TT * NN * HD * 4;
  float* agg = (float*)(ws + off); off += (size_t)TT * NN * HD * 4;
  unsigned* wfE1 = (unsigned*)(ws + off); off += (size_t)4 * 5120 * 4;
  unsigned* wfE2 = (unsigned*)(ws + off); off += (size_t)4 * 2048 * 4;
  unsigned* wfN1 = (unsigned*)(ws + off); off += (size_t)4 * 4096 * 4;
  unsigned* wfN2 = (unsigned*)(ws + off); off += (size_t)4 * 2048 * 4;
  unsigned* wfD1 = (unsigned*)(ws + off); off += (size_t)2048 * 4;
  unsigned* wfD2 = (unsigned*)(ws + off); off += (size_t)512 * 4;

  auto conv = [&](const float* s, unsigned* d, int b, int r, int c, int KC, int NT) {
    int tot = b * KC * NT * 256;
    conv_wfrag<<<(tot + 255) / 256, 256, 0, stream>>>(s, d, b, r, c, KC, NT);
  };
  conv(edge_w1, wfE1, 4, 130, 64, 5, 4);
  conv(edge_w2, wfE2, 4,  64, 64, 2, 4);
  conv(node_w1, wfN1, 4, 128, 64, 4, 4);
  conv(node_w2, wfN2, 4,  64, 64, 2, 4);
  conv(dec_w1,  wfD1, 1,  64, 64, 2, 4);
  conv(dec_w2,  wfD2, 1,  64, 12, 2, 1);

  {
    int tot = TT * NN * HD;
    embed_kernel<<<(tot + 255) / 256, 256, 0, stream>>>(nodes, emb_w, emb_b, time_emb, h);
  }

  for (int l = 0; l < 4; ++l) {
    int nAgg = TT * NN * HD;
    zero_kernel<<<(nAgg + 255) / 256, 256, 0, stream>>>(agg, nAgg);
    edge_kernel<<<1024, 128, 0, stream>>>(h, edges, edge_attr, wfE1, wfE2,
                                          edge_b1, edge_b2, agg, l);
    node_kernel<<<320, 256, 0, stream>>>(h, agg, wfN1, wfN2, node_b1, node_b2, l);
  }

  dec_kernel<<<64, 256, 0, stream>>>(h, wfD1, wfD2, dec_b1, dec_b2, theta, (float*)d_out);
}
